// GraphAttentionNetwork_81853486727297
// MI455X (gfx1250) — matrix operand order repaired and checked by reading, not compiled, not run on hardware
//
#include <hip/hip_runtime.h>
#include <math.h>

typedef __attribute__((ext_vector_type(16))) __bf16 v16bf;
typedef __attribute__((ext_vector_type(8)))  float  v8f;

static constexpr float kNegSlope = 0.2f;
static constexpr float kEpsSm    = 1e-16f;
static constexpr float kEpsLn    = 1e-5f;

// ---------------------------------------------------------------------------
// Pack row-major W[K, NCOLS] (fp32) into WMMA B-fragment order (bf16).
// Fragment layout (V_WMMA_F32_16X16X32_BF16, wave32):
//   lane 0-15 : N = lane,     elems 0..7 = K 0..7,  elems 8..15 = K 16..23
//   lane 16-31: N = lane-16,  elems 0..7 = K 8..15, elems 8..15 = K 24..31
// Flat index: ((kt*NT + nt)*32 + lane)*16 + j
// ---------------------------------------------------------------------------
__global__ void pack_w_kernel(const float* __restrict__ W, __bf16* __restrict__ Wf,
                              int KTOT, int NCOLS) {
  int idx = blockIdx.x * 256 + threadIdx.x;
  int total = KTOT * NCOLS;
  if (idx >= total) return;
  int j    = idx & 15;
  int lane = (idx >> 4) & 31;
  int t    = idx >> 9;
  int NT   = NCOLS >> 4;
  int kt   = t / NT;
  int nt   = t - kt * NT;
  int k = kt * 32 + ((lane >> 4) << 3) + (j < 8 ? j : j + 8);
  int n = nt * 16 + (lane & 15);
  Wf[idx] = (__bf16)W[(size_t)k * NCOLS + n];
}

// ---------------------------------------------------------------------------
// GEMM: H[M,NCOLS] = A[M,KTOT] @ W  via v_wmma_f32_16x16x32_bf16.
// Each wave computes a 16 x (16*NBLK) strip: one A fragment per K-step feeds
// NBLK independent WMMAs (independent accumulators -> no RAW hazard NOPs,
// 3 vmem instructions per WMMA instead of 6, 4x less L2 re-read of A).
// A-fragment layout (16-bit A 16x32):
//   lane 0-15 : M = lane,    elems 0..7 = K 0..7,  elems 8..15 = K 16..23
//   lane 16-31: M = lane-16, elems 0..7 = K 8..15, elems 8..15 = K 24..31
// ---------------------------------------------------------------------------
template <int KTOT, int NCOLS, int NBLK>
__global__ __launch_bounds__(256) void gemm_wmma_kernel(
    const float* __restrict__ A, const __bf16* __restrict__ Wf,
    float* __restrict__ H, int M) {
  constexpr int NT     = NCOLS / 16;   // 16-col tiles
  constexpr int NGROUP = NT / NBLK;    // strips per row of tiles
  int wid  = (blockIdx.x * 256 + (int)threadIdx.x) >> 5;
  int lane = threadIdx.x & 31;
  int tilesM = (M + 15) >> 4;
  if (wid >= tilesM * NGROUP) return;
  int tileM = wid / NGROUP;
  int grpN  = wid - tileM * NGROUP;

  int row = tileM * 16 + (lane & 15);
  if (row >= M) row = M - 1;          // clamp (M is a multiple of 16 here)
  int k0 = (lane >> 4) << 3;          // 0 or 8
  const v16bf* Wt = (const v16bf*)Wf;

  v8f acc[NBLK];
#pragma unroll
  for (int b = 0; b < NBLK; ++b) acc[b] = (v8f){};

#pragma unroll
  for (int kt = 0; kt < KTOT / 32; ++kt) {
    // ---- A fragment (shared by all NBLK WMMAs this K-step) ----
    const float* ap = A + (size_t)row * KTOT + kt * 32 + k0;
    float4 a0 = *(const float4*)(ap);
    float4 a1 = *(const float4*)(ap + 4);
    float4 a2 = *(const float4*)(ap + 16);
    float4 a3 = *(const float4*)(ap + 20);
    float af[16] = {a0.x, a0.y, a0.z, a0.w, a1.x, a1.y, a1.z, a1.w,
                    a2.x, a2.y, a2.z, a2.w, a3.x, a3.y, a3.z, a3.w};
    v16bf a;
#pragma unroll
    for (int i = 0; i < 16; ++i) a[i] = (__bf16)af[i];

    // ---- NBLK independent B fragments / accumulators ----
#pragma unroll
    for (int b = 0; b < NBLK; ++b) {
      v16bf bf = Wt[(size_t)(kt * NT + grpN * NBLK + b) * 32 + lane];
      acc[b] = __builtin_amdgcn_wmma_f32_16x16x32_bf16(
          false, a, false, bf, (short)0, acc[b], false, false);
    }
  }

  // C/D layout: lanes 0-15 N=lane, VGPR r -> M=r; lanes 16-31 -> M=8+r
  int mbase = tileM * 16 + ((lane >> 4) << 3);
#pragma unroll
  for (int b = 0; b < NBLK; ++b) {
    int ncol = (grpN * NBLK + b) * 16 + (lane & 15);
#pragma unroll
    for (int r = 0; r < 8; ++r) {
      int m = mbase + r;
      if (m < M) H[(size_t)m * NCOLS + ncol] = acc[b][r];
    }
  }
}

// ---------------------------------------------------------------------------
// Per-node attention scores: s[n,h] = sum_c H[n,h,c] * a[h,c]
// ---------------------------------------------------------------------------
template <int HEADS, int CH>
__global__ void scores_kernel(const float* __restrict__ H,
                              const float* __restrict__ asrc,
                              const float* __restrict__ adst,
                              float* __restrict__ ssrc, float* __restrict__ sdst,
                              int n) {
  int idx = blockIdx.x * 256 + threadIdx.x;
  if (idx >= n * HEADS) return;
  int node = idx / HEADS;
  int head = idx - node * HEADS;
  const float* hp = H + (size_t)node * (HEADS * CH) + head * CH;
  const float* ap = asrc + head * CH;
  const float* dp = adst + head * CH;
  float s1 = 0.f, s2 = 0.f;
#pragma unroll 4
  for (int c = 0; c < CH; ++c) {
    float hv = hp[c];
    s1 += hv * ap[c];
    s2 += hv * dp[c];
  }
  ssrc[idx] = s1;
  sdst[idx] = s2;
}

__global__ void zero_f32_kernel(float* __restrict__ p, long n) {
  long i = (long)blockIdx.x * 256 + threadIdx.x;
  if (i < n) p[i] = 0.f;
}

// ---------------------------------------------------------------------------
// Edge pass 1: a_exp = exp(leaky_relu(s_src[row]+s_dst[col]) * ew);
//              denom[col] += a_exp   (f32 atomics, L2-resident)
// ---------------------------------------------------------------------------
template <int HEADS>
__global__ void edge_softmax_num_kernel(const int* __restrict__ row,
                                        const int* __restrict__ col,
                                        const float* __restrict__ ew,
                                        const float* __restrict__ ssrc,
                                        const float* __restrict__ sdst,
                                        float* __restrict__ aexp,
                                        float* __restrict__ denom, int E) {
  int e = blockIdx.x * 256 + threadIdx.x;
  if (e >= E) return;
  int r = row[e], c = col[e];
  float w = ew[e];
#pragma unroll
  for (int h = 0; h < HEADS; ++h) {
    float a = ssrc[r * HEADS + h] + sdst[c * HEADS + h];
    a = (a > 0.f) ? a : kNegSlope * a;
    float ex = expf(a * w);
    aexp[(size_t)e * HEADS + h] = ex;
    atomicAdd(&denom[c * HEADS + h], ex);
  }
}

// ---------------------------------------------------------------------------
// Edge pass 2: agg[row] += (aexp / (denom[col]+eps)) * h[col]
// D/4 lanes per edge, float4 gather + 4 f32 atomics (L2-resident).
// ---------------------------------------------------------------------------
template <int HEADS, int CH>
__global__ void edge_aggregate_kernel(const int* __restrict__ row,
                                      const int* __restrict__ col,
                                      const float* __restrict__ aexp,
                                      const float* __restrict__ denom,
                                      const float* __restrict__ H,
                                      float* __restrict__ out, int E) {
  constexpr int D   = HEADS * CH;
  constexpr int TPE = D / 4;
  long gid = (long)blockIdx.x * 256 + threadIdx.x;
  int e = (int)(gid / TPE);
  if (e >= E) return;
  int t = (int)(gid - (long)e * TPE);
  int cbase = t * 4;
  int head  = cbase / CH;
  int r = row[e], c = col[e];
  float alpha = aexp[(size_t)e * HEADS + head] /
                (denom[c * HEADS + head] + kEpsSm);
  float4 hv = *(const float4*)(H + (size_t)c * D + cbase);
  float* op = out + (size_t)r * D + cbase;
  atomicAdd(op + 0, alpha * hv.x);
  atomicAdd(op + 1, alpha * hv.y);
  atomicAdd(op + 2, alpha * hv.z);
  atomicAdd(op + 3, alpha * hv.w);
}

// ---------------------------------------------------------------------------
// Epilogue: y = agg + bias; LayerNorm over D; optional ELU. One wave per node.
// ---------------------------------------------------------------------------
template <int D, bool ELU>
__global__ __launch_bounds__(256) void ln_kernel(const float* __restrict__ agg,
                                                 const float* __restrict__ bias,
                                                 const float* __restrict__ g,
                                                 const float* __restrict__ be,
                                                 float* __restrict__ out, int n) {
  constexpr int PER = D / 32;
  int wid  = (blockIdx.x * 256 + (int)threadIdx.x) >> 5;
  int lane = threadIdx.x & 31;
  if (wid >= n) return;
  const float* ap = agg + (size_t)wid * D;
  float v[PER];
  float mu = 0.f;
#pragma unroll
  for (int i = 0; i < PER; ++i) {
    int ch = lane + i * 32;
    v[i] = ap[ch] + bias[ch];
    mu += v[i];
  }
#pragma unroll
  for (int off = 16; off > 0; off >>= 1) mu += __shfl_xor(mu, off, 32);
  mu *= (1.f / D);
  float var = 0.f;
#pragma unroll
  for (int i = 0; i < PER; ++i) {
    float d = v[i] - mu;
    var += d * d;
  }
#pragma unroll
  for (int off = 16; off > 0; off >>= 1) var += __shfl_xor(var, off, 32);
  var *= (1.f / D);
  float rs = rsqrtf(var + kEpsLn);
  float* op = out + (size_t)wid * D;
#pragma unroll
  for (int i = 0; i < PER; ++i) {
    int ch = lane + i * 32;
    float y = (v[i] - mu) * rs * g[ch] + be[ch];
    if (ELU) y = (y > 0.f) ? y : expm1f(y);
    op[ch] = y;
  }
}

// ---------------------------------------------------------------------------
static inline int cdiv(long a, long b) { return (int)((a + b - 1) / b); }

extern "C" void kernel_launch(void* const* d_in, const int* in_sizes, int n_in,
                              void* d_out, int out_size, void* d_ws, size_t ws_size,
                              hipStream_t stream) {
  const float* x   = (const float*)d_in[0];
  const int*   ei  = (const int*)d_in[1];
  const float* ew  = (const float*)d_in[2];
  const float* W1  = (const float*)d_in[3];
  const float* as1 = (const float*)d_in[4];
  const float* ad1 = (const float*)d_in[5];
  const float* b1  = (const float*)d_in[6];
  const float* g1  = (const float*)d_in[7];
  const float* be1 = (const float*)d_in[8];
  const float* W2  = (const float*)d_in[9];
  const float* as2 = (const float*)d_in[10];
  const float* ad2 = (const float*)d_in[11];
  const float* b2  = (const float*)d_in[12];
  const float* g2  = (const float*)d_in[13];
  const float* be2 = (const float*)d_in[14];
  const float* W3  = (const float*)d_in[15];
  const float* as3 = (const float*)d_in[16];
  const float* ad3 = (const float*)d_in[17];
  const float* b3  = (const float*)d_in[18];
  const float* g3  = (const float*)d_in[19];
  const float* be3 = (const float*)d_in[20];

  const int N = in_sizes[0] / 384;
  const int E = in_sizes[2];
  const int* row = ei;
  const int* col = ei + E;

  // -------- workspace layout --------
  char* ws = (char*)d_ws;
  __bf16* w1f = (__bf16*)ws;               // 384*256 bf16
  __bf16* w2f = w1f + 384 * 256;           // 256*256 bf16
  __bf16* w3f = w2f + 256 * 256;           // 256*128 bf16
  float* agg   = (float*)(ws + 393216);    // N*256 f32 (offset 384 KiB, aligned)
  float* hbuf  = agg  + (size_t)N * 256;   // N*256 f32
  float* ssrc  = hbuf + (size_t)N * 256;   // N*4
  float* sdst  = ssrc + (size_t)N * 4;     // N*4
  float* denom = sdst + (size_t)N * 4;     // N*4
  float* aexp  = denom + (size_t)N * 4;    // E*4

  // -------- pack weights to bf16 WMMA fragments --------
  pack_w_kernel<<<cdiv(384 * 256, 256), 256, 0, stream>>>(W1, w1f, 384, 256);
  pack_w_kernel<<<cdiv(256 * 256, 256), 256, 0, stream>>>(W2, w2f, 256, 256);
  pack_w_kernel<<<cdiv(256 * 128, 256), 256, 0, stream>>>(W3, w3f, 256, 128);

  const int tilesM = (N + 15) / 16;

  // ================= Layer 1 (384 -> 4x64, concat) =================
  gemm_wmma_kernel<384, 256, 4>
      <<<cdiv((long)tilesM * 4 * 32, 256), 256, 0, stream>>>(x, w1f, hbuf, N);
  scores_kernel<4, 64><<<cdiv((long)N * 4, 256), 256, 0, stream>>>(
      hbuf, as1, ad1, ssrc, sdst, N);
  zero_f32_kernel<<<cdiv((long)N * 4, 256), 256, 0, stream>>>(denom, (long)N * 4);
  zero_f32_kernel<<<cdiv((long)N * 256, 256), 256, 0, stream>>>(agg, (long)N * 256);
  edge_softmax_num_kernel<4><<<cdiv(E, 256), 256, 0, stream>>>(
      row, col, ew, ssrc, sdst, aexp, denom, E);
  edge_aggregate_kernel<4, 64><<<cdiv((long)E * 64, 256), 256, 0, stream>>>(
      row, col, aexp, denom, hbuf, agg, E);
  ln_kernel<256, true><<<cdiv((long)N * 32, 256), 256, 0, stream>>>(
      agg, b1, g1, be1, agg, N);

  // ================= Layer 2 (256 -> 4x64, concat) =================
  gemm_wmma_kernel<256, 256, 4>
      <<<cdiv((long)tilesM * 4 * 32, 256), 256, 0, stream>>>(agg, w2f, hbuf, N);
  scores_kernel<4, 64><<<cdiv((long)N * 4, 256), 256, 0, stream>>>(
      hbuf, as2, ad2, ssrc, sdst, N);
  zero_f32_kernel<<<cdiv((long)N * 4, 256), 256, 0, stream>>>(denom, (long)N * 4);
  zero_f32_kernel<<<cdiv((long)N * 256, 256), 256, 0, stream>>>(agg, (long)N * 256);
  edge_softmax_num_kernel<4><<<cdiv(E, 256), 256, 0, stream>>>(
      row, col, ew, ssrc, sdst, aexp, denom, E);
  edge_aggregate_kernel<4, 64><<<cdiv((long)E * 64, 256), 256, 0, stream>>>(
      row, col, aexp, denom, hbuf, agg, E);
  ln_kernel<256, true><<<cdiv((long)N * 32, 256), 256, 0, stream>>>(
      agg, b2, g2, be2, agg, N);

  // ================= Layer 3 (256 -> 128, 1 head, mean==identity) ==
  gemm_wmma_kernel<256, 128, 4>
      <<<cdiv((long)tilesM * 2 * 32, 256), 256, 0, stream>>>(agg, w3f, hbuf, N);
  scores_kernel<1, 128><<<cdiv(N, 256), 256, 0, stream>>>(
      hbuf, as3, ad3, ssrc, sdst, N);
  zero_f32_kernel<<<cdiv(N, 256), 256, 0, stream>>>(denom, (long)N);
  zero_f32_kernel<<<cdiv((long)N * 128, 256), 256, 0, stream>>>(agg, (long)N * 128);
  edge_softmax_num_kernel<1><<<cdiv(E, 256), 256, 0, stream>>>(
      row, col, ew, ssrc, sdst, aexp, denom, E);
  edge_aggregate_kernel<1, 128><<<cdiv((long)E * 32, 256), 256, 0, stream>>>(
      row, col, aexp, denom, hbuf, agg, E);
  ln_kernel<128, false><<<cdiv((long)N * 32, 256), 256, 0, stream>>>(
      agg, b3, g3, be3, (float*)d_out, N);
}
